// GAT_30562987278781
// MI455X (gfx1250) — matrix operand order repaired and checked by reading, not compiled, not run on hardware
//
#include <hip/hip_runtime.h>
#include <math.h>

typedef __attribute__((ext_vector_type(16))) _Float16 v16h;
typedef __attribute__((ext_vector_type(8)))  _Float16 v8h;
typedef __attribute__((ext_vector_type(8)))  float    v8f;

#define NEG_SLOPE 0.2f

// ---------------------------------------------------------------- utilities
__global__ void fill_kernel(float* __restrict__ p, float v, int n) {
  int i = blockIdx.x * blockDim.x + threadIdx.x;
  if (i < n) p[i] = v;
}

__global__ void cvt_f16_kernel(const float* __restrict__ in, _Float16* __restrict__ out, int n) {
  int i = blockIdx.x * blockDim.x + threadIdx.x;
  if (i < n) out[i] = (_Float16)in[i];
}

// W is [K, NC] row-major f32; WT is [NC, K] row-major f16 (so B columns are contiguous)
__global__ void transpose_f16_kernel(const float* __restrict__ W, _Float16* __restrict__ WT,
                                     int K, int NC) {
  int i = blockIdx.x * blockDim.x + threadIdx.x;
  if (i >= K * NC) return;
  int k = i / NC, n = i % NC;
  WT[(size_t)n * K + k] = (_Float16)W[i];
}

// ------------------------------------------------- WMMA GEMM: C = A(f16) * B(f16), f32 acc
// A: [M,K] row-major f16.  BT: [NC,K] row-major f16 (B transposed).  C: [M,NC] f32.
// One wave computes one 16x16 tile: blockIdx.x = row tile, wave-in-block = col tile.
// K must be a multiple of 32, M a multiple of 16 (100000 = 6250*16).
__global__ void gemm_wmma_kernel(const _Float16* __restrict__ A,
                                 const _Float16* __restrict__ BT,
                                 float* __restrict__ C,
                                 int M, int K, int NC) {
  const int lane = threadIdx.x & 31;
  const int wave = threadIdx.x >> 5;
  const int row0 = blockIdx.x * 16;
  const int n0   = wave * 16;
  const int l15  = lane & 15;
  const int sel  = (lane >> 4) & 1;   // lane-half selects K sub-range (ISA 7.12.2)

  const _Float16* arow = A  + (size_t)(row0 + l15) * K;
  const _Float16* brow = BT + (size_t)(n0   + l15) * K;

  v8f acc = {};
  for (int k0 = 0; k0 < K; k0 += 32) {
    // A 16x32 f16 fragment: halves 0..7 -> K = k0 + sel*16 + t ; halves 8..15 -> K = k0+sel*16+8+t
    v8h alo = *(const v8h*)(arow + k0 + sel * 16);
    v8h ahi = *(const v8h*)(arow + k0 + sel * 16 + 8);
    // B 32x16 f16 fragment: halves t -> K = k0 + sel*16 + t, col = n0 + l15
    v16h b = *(const v16h*)(brow + k0 + sel * 16);
    v16h a;
#pragma unroll
    for (int i = 0; i < 8; ++i) { a[i] = alo[i]; a[8 + i] = ahi[i]; }
    acc = __builtin_amdgcn_wmma_f32_16x16x32_f16(false, a, false, b,
                                                 (short)0, acc, false, false);
  }
  // C/D layout: VGPR r -> row = row0 + sel*8 + r ; col = n0 + l15
  const int n = n0 + l15;
#pragma unroll
  for (int r = 0; r < 8; ++r) {
    C[(size_t)(row0 + sel * 8 + r) * NC + n] = acc[r];
  }
}

// ---------------------------------------------- alpha_s / alpha_d per (node, head)
__global__ void alpha_kernel(const float* __restrict__ Hb,
                             const float* __restrict__ a_src,
                             const float* __restrict__ a_dst,
                             float* __restrict__ as, float* __restrict__ ad,
                             int n, int H, int C) {
  int t = blockIdx.x * blockDim.x + threadIdx.x;
  if (t >= n * H) return;
  int node = t / H, h = t % H;
  const float* hp = Hb + (size_t)node * H * C + (size_t)h * C;
  float ss = 0.f, sd = 0.f;
  for (int c = 0; c < C; ++c) {
    float v = hp[c];
    ss += v * a_src[h * C + c];
    sd += v * a_dst[h * C + c];
  }
  as[t] = ss;
  ad[t] = sd;
}

// f32 atomic max via int-max / uint-min bit trick (valid for non-NaN, init -inf)
__device__ __forceinline__ void atomicMaxF(float* addr, float v) {
  if (v >= 0.f) atomicMax((int*)addr, __float_as_int(v));
  else          atomicMin((unsigned int*)addr, __float_as_uint(v));
}

// edges [0,E) come from edge_index; edges [E, E+n) are self-loops
__global__ void edge_max_kernel(const float* __restrict__ as, const float* __restrict__ ad,
                                const int* __restrict__ src, const int* __restrict__ dst,
                                float* __restrict__ ebuf, float* __restrict__ mbuf,
                                int E, int n, int H) {
  int e = blockIdx.x * blockDim.x + threadIdx.x;
  if (e >= E + n) return;
  int s, d;
  if (e < E) { s = src[e]; d = dst[e]; } else { s = d = e - E; }
  for (int h = 0; h < H; ++h) {
    float v = as[s * H + h] + ad[d * H + h];
    v = (v > 0.f) ? v : NEG_SLOPE * v;           // leaky_relu
    ebuf[(size_t)e * H + h] = v;
    atomicMaxF(&mbuf[d * H + h], v);
  }
}

__global__ void edge_expsum_kernel(float* __restrict__ ebuf, const float* __restrict__ mbuf,
                                   float* __restrict__ denom, const int* __restrict__ dst,
                                   int E, int n, int H) {
  int e = blockIdx.x * blockDim.x + threadIdx.x;
  if (e >= E + n) return;
  int d = (e < E) ? dst[e] : (e - E);
  for (int h = 0; h < H; ++h) {
    float mm = mbuf[d * H + h];
    if (!(mm >= -3.0e38f && mm <= 3.0e38f)) mm = 0.f;   // match reference finite-guard
    float ex = expf(ebuf[(size_t)e * H + h] - mm);
    ebuf[(size_t)e * H + h] = ex;
    atomicAdd(&denom[d * H + h], ex);
  }
}

// wave-per-edge scatter: lane = channel (C==32) -> coalesced gathers and atomics
__global__ void edge_msg_kernel(const float* __restrict__ Hb, const float* __restrict__ ebuf,
                                const float* __restrict__ denom,
                                const int* __restrict__ src, const int* __restrict__ dst,
                                float* __restrict__ out, int E, int n, int H) {
  int gid  = blockIdx.x * blockDim.x + threadIdx.x;
  int e    = gid >> 5;
  int lane = gid & 31;
  if (e >= E + n) return;
  int s, d;
  if (e < E) { s = src[e]; d = dst[e]; } else { s = d = e - E; }
  const float* hs = Hb + (size_t)s * H * 32;
  float* od = out + (size_t)d * H * 32;
  for (int k = 0; k < H; ++k) {
    float alpha = ebuf[(size_t)e * H + k] / (denom[d * H + k] + 1e-16f);
    atomicAdd(&od[k * 32 + lane], hs[k * 32 + lane] * alpha);
  }
}

// out = elu(in + bias), written as f16 (feeds next WMMA GEMM)
__global__ void bias_elu_f16_kernel(const float* __restrict__ in, const float* __restrict__ bias,
                                    _Float16* __restrict__ out, int n, int F) {
  int i = blockIdx.x * blockDim.x + threadIdx.x;
  if (i >= n * F) return;
  float v = in[i] + bias[i % F];
  v = (v > 0.f) ? v : expm1f(v);
  out[i] = (_Float16)v;
}

// final elu + global mean-pool accumulation
__global__ void bias_elu_pool_kernel(const float* __restrict__ in, const float* __restrict__ bias,
                                     const int* __restrict__ batch,
                                     float* __restrict__ gsum, float* __restrict__ gcnt,
                                     int n, int C) {
  int i = blockIdx.x * blockDim.x + threadIdx.x;
  if (i >= n * C) return;
  int node = i / C, c = i % C;
  float v = in[i] + bias[c];
  v = (v > 0.f) ? v : expm1f(v);
  int g = batch[node];
  atomicAdd(&gsum[g * C + c], v);
  if (c == 0) atomicAdd(&gcnt[g], 1.0f);
}

__global__ void classifier_kernel(const float* __restrict__ gsum, const float* __restrict__ gcnt,
                                  const float* __restrict__ Wlin, const float* __restrict__ blin,
                                  float* __restrict__ out, int G, int C, int K) {
  int i = blockIdx.x * blockDim.x + threadIdx.x;
  if (i >= G * K) return;
  int g = i / K, k = i % K;
  float cnt = gcnt[g];
  cnt = (cnt > 1.f) ? cnt : 1.f;
  float acc = blin[k];
  for (int c = 0; c < C; ++c) acc += (gsum[g * C + c] / cnt) * Wlin[c * K + k];
  out[i] = acc;
}

// ---------------------------------------------------------------- launch
extern "C" void kernel_launch(void* const* d_in, const int* in_sizes, int n_in,
                              void* d_out, int out_size, void* d_ws, size_t ws_size,
                              hipStream_t stream) {
  const int N = 100000, E = 1600000, INC = 128, HID = 32, HEADS = 4, G = 256, NCLS = 10;
  const int F1 = HEADS * HID;   // 128
  const int ET = E + N;         // edges + self-loops

  const float* x     = (const float*)d_in[0];
  const int*   ei    = (const int*)d_in[1];
  const int*   batch = (const int*)d_in[2];
  const float* W1    = (const float*)d_in[3];
  const float* as1w  = (const float*)d_in[4];
  const float* ad1w  = (const float*)d_in[5];
  const float* b1    = (const float*)d_in[6];
  const float* W2    = (const float*)d_in[7];
  const float* as2w  = (const float*)d_in[8];
  const float* ad2w  = (const float*)d_in[9];
  const float* b2    = (const float*)d_in[10];
  const float* Wlin  = (const float*)d_in[11];
  const float* blin  = (const float*)d_in[12];
  const int* src = ei;
  const int* dst = ei + E;

  // workspace carve-up (256B aligned regions)
  char* ws = (char*)d_ws;
  size_t off = 0;
  auto alloc = [&](size_t bytes) -> char* {
    char* p = ws + off;
    off = (off + bytes + 255) & ~(size_t)255;
    return p;
  };
  _Float16* xh   = (_Float16*)alloc((size_t)N * F1 * 2);   // layer-1 GEMM input; reused for layer-2
  _Float16* w1t  = (_Float16*)alloc((size_t)F1 * INC * 2);
  _Float16* w2t  = (_Float16*)alloc((size_t)HID * F1 * 2);
  float*    h1   = (float*)alloc((size_t)N * F1 * 4);      // reused for all layer-2 temporaries
  float*    out1 = (float*)alloc((size_t)N * F1 * 4);
  float*    ebuf = (float*)alloc((size_t)ET * HEADS * 4);  // reused for layer-2 (needs ET*1)
  float*    as1  = (float*)alloc((size_t)N * HEADS * 4);
  float*    ad1  = (float*)alloc((size_t)N * HEADS * 4);
  float*    m1   = (float*)alloc((size_t)N * HEADS * 4);
  float*    dn1  = (float*)alloc((size_t)N * HEADS * 4);
  float*    gsum = (float*)alloc((size_t)G * HID * 4);
  float*    gcnt = (float*)alloc((size_t)G * 4);
  // layer-2 temporaries live inside freed h1 region (N*(32+32+4) < N*128 floats)
  float* h2   = h1;
  float* out2 = h1 + (size_t)N * HID;
  float* as2  = out2 + (size_t)N * HID;
  float* ad2  = as2 + N;
  float* m2   = ad2 + N;
  float* dn2  = m2 + N;

  const int TB = 256;
  auto cdiv = [](long long a, long long b) { return (int)((a + b - 1) / b); };

  // precision conversion + weight transposes
  cvt_f16_kernel<<<cdiv((long long)N * F1, TB), TB, 0, stream>>>(x, xh, N * F1);
  transpose_f16_kernel<<<cdiv(INC * F1, TB), TB, 0, stream>>>(W1, w1t, INC, F1);
  transpose_f16_kernel<<<cdiv(F1 * HID, TB), TB, 0, stream>>>(W2, w2t, F1, HID);

  // ---------------- layer 1 (H=4, C=32) ----------------
  gemm_wmma_kernel<<<N / 16, 8 * 32, 0, stream>>>(xh, w1t, h1, N, INC, F1);
  alpha_kernel<<<cdiv((long long)N * HEADS, TB), TB, 0, stream>>>(h1, as1w, ad1w, as1, ad1, N, HEADS, HID);
  fill_kernel<<<cdiv((long long)N * HEADS, TB), TB, 0, stream>>>(m1, -INFINITY, N * HEADS);
  fill_kernel<<<cdiv((long long)N * HEADS, TB), TB, 0, stream>>>(dn1, 0.f, N * HEADS);
  fill_kernel<<<cdiv((long long)N * F1, TB), TB, 0, stream>>>(out1, 0.f, N * F1);
  edge_max_kernel<<<cdiv(ET, TB), TB, 0, stream>>>(as1, ad1, src, dst, ebuf, m1, E, N, HEADS);
  edge_expsum_kernel<<<cdiv(ET, TB), TB, 0, stream>>>(ebuf, m1, dn1, dst, E, N, HEADS);
  edge_msg_kernel<<<cdiv((long long)ET * 32, TB), TB, 0, stream>>>(h1, ebuf, dn1, src, dst, out1, E, N, HEADS);
  bias_elu_f16_kernel<<<cdiv((long long)N * F1, TB), TB, 0, stream>>>(out1, b1, xh, N, F1);

  // ---------------- layer 2 (H=1, C=32) ----------------
  gemm_wmma_kernel<<<N / 16, 2 * 32, 0, stream>>>(xh, w2t, h2, N, F1, HID);
  alpha_kernel<<<cdiv(N, TB), TB, 0, stream>>>(h2, as2w, ad2w, as2, ad2, N, 1, HID);
  fill_kernel<<<cdiv(N, TB), TB, 0, stream>>>(m2, -INFINITY, N);
  fill_kernel<<<cdiv(N, TB), TB, 0, stream>>>(dn2, 0.f, N);
  fill_kernel<<<cdiv((long long)N * HID, TB), TB, 0, stream>>>(out2, 0.f, N * HID);
  edge_max_kernel<<<cdiv(ET, TB), TB, 0, stream>>>(as2, ad2, src, dst, ebuf, m2, E, N, 1);
  edge_expsum_kernel<<<cdiv(ET, TB), TB, 0, stream>>>(ebuf, m2, dn2, dst, E, N, 1);
  edge_msg_kernel<<<cdiv((long long)ET * 32, TB), TB, 0, stream>>>(h2, ebuf, dn2, src, dst, out2, E, N, 1);

  // ---------------- pool + classifier ----------------
  fill_kernel<<<cdiv(G * HID, TB), TB, 0, stream>>>(gsum, 0.f, G * HID);
  fill_kernel<<<1, TB, 0, stream>>>(gcnt, 0.f, G);
  bias_elu_pool_kernel<<<cdiv((long long)N * HID, TB), TB, 0, stream>>>(out2, b2, batch, gsum, gcnt, N, HID);
  classifier_kernel<<<cdiv(G * NCLS, TB), TB, 0, stream>>>(gsum, gcnt, Wlin, blin, (float*)d_out, G, HID, NCLS);
}
